// DGCNN2_56796647522639
// MI455X (gfx1250) — compile-verified, matrix operand order checked by reading, not run on hardware
//
#include <hip/hip_runtime.h>
#include <hip/hip_bf16.h>
#include <math.h>

// ---------------------------------------------------------------------------
// DGCNN inference for MI455X (gfx1250, wave32, WMMA).
//  - kNN distances: V_WMMA_F32_16X16X4_F32 Gram tiles, full 16x4096 distance
//    stripe resident in LDS, fused top-20 select.
//  - EdgeConv MLP: bf16 V_WMMA_F32_16X16X32_BF16, msg/W1/W2 staged in LDS
//    (weights pre-swizzled so every fragment is two 16B loads),
//    fused GEMM1 -> BN/ReLU -> GEMM2 -> max-over-20 aggregation.
//  - Head MLP: bf16 WMMA GEMMs, wave = 16x64 tile, swizzled weights in ws.
// ---------------------------------------------------------------------------

typedef __attribute__((ext_vector_type(2)))  float  v2f;
typedef __attribute__((ext_vector_type(8)))  float  v8f;
typedef __attribute__((ext_vector_type(8)))  __bf16 v8bf;
typedef __attribute__((ext_vector_type(16))) __bf16 v16bf;

#define BATCH 8
#define NPT   4096
#define KNN   20
#define BN_S  0.99999500003749981f  // 1/sqrt(1+1e-5)

// Assemble a 16-element bf16 fragment from two contiguous 16B runs.
__device__ __forceinline__ v16bf ldfrag(const __bf16* p0, const __bf16* p1) {
  v8bf lo = *(const v8bf*)p0;
  v8bf hi = *(const v8bf*)p1;
  return __builtin_shufflevector(lo, hi, 0, 1, 2, 3, 4, 5, 6, 7,
                                 8, 9, 10, 11, 12, 13, 14, 15);
}

// ---------------------------------------------------------------------------
__global__ __launch_bounds__(256) void k_concat(const float* __restrict__ x,
                                                const float* __restrict__ pos,
                                                float* __restrict__ F0f,
                                                __bf16* __restrict__ F0h) {
  int t = blockIdx.x * blockDim.x + threadIdx.x;
  if (t >= BATCH * NPT) return;
  #pragma unroll
  for (int c = 0; c < 8; ++c) {
    float v = (c < 3) ? x[(size_t)t * 3 + c]
            : (c < 6) ? pos[(size_t)t * 3 + (c - 3)] : 0.f;
    F0f[(size_t)t * 8 + c] = v;
    F0h[(size_t)t * 8 + c] = (__bf16)v;
  }
}

// Swizzle f32 weights [K,N] -> bf16 B-fragment-contiguous layout:
// Ws[(ks*N + n)*32 + j], j<16 half0 / j>=16 half1, so a lane's fragment is
// two contiguous 16B loads.
__global__ __launch_bounds__(256) void k_swz(const float* __restrict__ W,
                                             __bf16* __restrict__ Ws,
                                             int Kdim, int Nc) {
  int t = blockIdx.x * blockDim.x + threadIdx.x;
  if (t >= Kdim * Nc) return;
  int ks = t / (Nc * 32);
  int rem = t % (Nc * 32);
  int n = rem >> 5;
  int j = rem & 31;
  int half = j >> 4, jj = j & 15;
  int k = ks * 32 + ((jj < 8) ? half * 8 + jj : 16 + half * 8 + (jj - 8));
  Ws[t] = (__bf16)W[k * Nc + n];
}

// ---------------------------------------------------------------------------
// kNN: one block = (batch b, 16 rows). Distance stripe 16x4096 in LDS via
// f32 WMMA Gram tiles, then per-row insertion top-20.
// ---------------------------------------------------------------------------
template <int CIN, int CP>
__global__ __launch_bounds__(256) void k_knn(const float* __restrict__ F,
                                             int* __restrict__ idxg) {
  extern __shared__ float sm[];
  float* dist = sm;               // 16*4096 f32
  float* sqj  = sm + 16 * NPT;    // 4096 f32
  float* sqi  = sqj + NPT;        // 16 f32

  const int b = blockIdx.y;
  const int rbase = blockIdx.x * 16;
  const int tid = threadIdx.x;
  const int lane = tid & 31, wave = tid >> 5;
  const float* Fb = F + (size_t)b * NPT * CP;

  for (int j = tid; j < NPT; j += 256) {
    float s = 0.f;
    #pragma unroll
    for (int c = 0; c < CIN; ++c) { float v = Fb[(size_t)j * CP + c]; s += v * v; }
    sqj[j] = s;
  }
  __syncthreads();
  if (tid < 16) sqi[tid] = sqj[rbase + tid];
  __syncthreads();

  const int m15 = lane & 15, kb = (lane >> 4) * 2, mb8 = (lane >> 4) * 8;
  float sqir[8];
  #pragma unroll
  for (int r = 0; r < 8; ++r) sqir[r] = sqi[mb8 + r];

  v2f afr[CP / 4];
  #pragma unroll
  for (int s = 0; s < CP / 4; ++s) {
    const float* rp = Fb + (size_t)(rbase + m15) * CP + 4 * s + kb;
    afr[s].x = rp[0];
    afr[s].y = rp[1];
  }
  for (int ct = wave; ct < NPT / 16; ct += 8) {
    int cb = ct * 16;
    v8f acc = {};
    #pragma unroll
    for (int s = 0; s < CP / 4; ++s) {
      const float* cp = Fb + (size_t)(cb + m15) * CP + 4 * s + kb;
      v2f bfr;
      bfr.x = cp[0];
      bfr.y = cp[1];
      acc = __builtin_amdgcn_wmma_f32_16x16x4_f32(false, afr[s], false, bfr,
                                                  (short)0, acc, false, false);
    }
    float sj = sqj[cb + m15];
    #pragma unroll
    for (int r = 0; r < 8; ++r)
      dist[(mb8 + r) * NPT + cb + m15] = sqir[r] + sj - 2.f * acc[r];
  }
  __syncthreads();

  if (tid < 16) {
    float bd[KNN];
    int bi[KNN];
    #pragma unroll
    for (int k = 0; k < KNN; ++k) { bd[k] = 3.0e38f; bi[k] = 0; }
    const float* drow = dist + tid * NPT;
    for (int j = 0; j < NPT; ++j) {
      float d = drow[j];
      if (d < bd[KNN - 1]) {
        int p = KNN - 1;
        while (p > 0 && bd[p - 1] > d) { bd[p] = bd[p - 1]; bi[p] = bi[p - 1]; --p; }
        bd[p] = d;
        bi[p] = j;
      }
    }
    int* op = idxg + ((size_t)b * NPT + rbase + tid) * KNN;
    #pragma unroll
    for (int k = 0; k < KNN; ++k) op[k] = bi[k];
  }
}

// ---------------------------------------------------------------------------
// EdgeConv: one block = (batch b, 16 points) = 320 edges = 20 M-tiles.
// msg [320,KP] bf16 in LDS, GEMM1(KPx64) -> BN/ReLU -> GEMM2(64x64) -> max.
// Weights staged in LDS in B-fragment-swizzled order.
// ---------------------------------------------------------------------------
template <int CIN, int FSTR, int KP>
__global__ __launch_bounds__(256) void k_edge(
    const __bf16* __restrict__ Fh, const int* __restrict__ idxg,
    const float* __restrict__ W1g, const float* __restrict__ b1g,
    const float* __restrict__ gg, const float* __restrict__ beg,
    const float* __restrict__ W2g, const float* __restrict__ b2g,
    float* __restrict__ Xf, __bf16* __restrict__ Xh,
    __bf16* __restrict__ Hcat, int hoff) {
  constexpr int NPTS = 16;
  constexpr int NE = NPTS * KNN;  // 320 edges
  extern __shared__ char smraw[];
  __bf16* msg = (__bf16*)smraw;              // NE*KP
  __bf16* w1  = msg + NE * KP;               // KP*64 (swizzled)
  __bf16* w2  = w1 + KP * 64;                // 64*64 (swizzled)
  __bf16* h1  = w2 + 64 * 64;                // 8 waves * 16*64
  float*  h2  = (float*)(h1 + 8 * 16 * 64);  // NE*64 f32
  float*  bb  = h2 + NE * 64;                // b1,g,be,b2 (4*64)
  int*   eidx = (int*)(bb + 4 * 64);         // NE

  const int b = blockIdx.y, pbase = blockIdx.x * NPTS;
  const int tid = threadIdx.x, lane = tid & 31, wave = tid >> 5;

  // stage swizzled weights + bias
  for (int t = tid; t < KP * 64; t += 256) {
    int ks = t / (64 * 32);
    int rem = t % (64 * 32);
    int n = rem >> 5;
    int j = rem & 31;
    int half = j >> 4, jj = j & 15;
    int k = ks * 32 + ((jj < 8) ? half * 8 + jj : 16 + half * 8 + (jj - 8));
    w1[t] = (__bf16)((k < 2 * CIN) ? W1g[k * 64 + n] : 0.f);
  }
  for (int t = tid; t < 64 * 64; t += 256) {
    int ks = t / (64 * 32);
    int rem = t % (64 * 32);
    int n = rem >> 5;
    int j = rem & 31;
    int half = j >> 4, jj = j & 15;
    int k = ks * 32 + ((jj < 8) ? half * 8 + jj : 16 + half * 8 + (jj - 8));
    w2[t] = (__bf16)W2g[k * 64 + n];
  }
  if (tid < 64) {
    bb[tid]       = b1g[tid];
    bb[64 + tid]  = gg[tid];
    bb[128 + tid] = beg[tid];
    bb[192 + tid] = b2g[tid];
  }
  for (int e = tid; e < NE; e += 256)
    eidx[e] = idxg[((size_t)b * NPT + pbase + e / KNN) * KNN + e % KNN];
  __syncthreads();

  // build bf16 messages [x_i, x_j - x_i] (zero-padded to KP)
  const __bf16* Fb = Fh + (size_t)b * NPT * FSTR;
  for (int t = tid; t < NE * KP; t += 256) {
    int e = t / KP, c = t % KP;
    int p = pbase + e / KNN;
    float v = 0.f;
    if (c < CIN) v = (float)Fb[(size_t)p * FSTR + c];
    else if (c < 2 * CIN) {
      int cc = c - CIN;
      int j = eidx[e];
      v = (float)Fb[(size_t)j * FSTR + cc] - (float)Fb[(size_t)p * FSTR + cc];
    }
    msg[t] = (__bf16)v;
  }
  __syncthreads();

  const int m15 = lane & 15, half = lane >> 4;
  const int mb8 = half * 8, koff = half * 8;
  for (int tile = wave; tile < NE / 16; tile += 8) {
    int ebase = tile * 16;
    v8f acc[4] = {};
    #pragma unroll
    for (int ks = 0; ks < KP / 32; ++ks) {
      const __bf16* ap = msg + (ebase + m15) * KP + ks * 32 + koff;
      v16bf a = ldfrag(ap, ap + 16);
      #pragma unroll
      for (int nt = 0; nt < 4; ++nt) {
        const __bf16* wp = w1 + (ks * 64 + nt * 16 + m15) * 32 + half * 16;
        v16bf bf = ldfrag(wp, wp + 8);
        acc[nt] = __builtin_amdgcn_wmma_f32_16x16x32_bf16(
            false, a, false, bf, (short)0, acc[nt], false, false);
      }
    }
    // BN(eval) + ReLU, transpose C-layout -> A-layout through LDS
    __bf16* h1w = h1 + wave * (16 * 64);
    #pragma unroll
    for (int nt = 0; nt < 4; ++nt) {
      int n = nt * 16 + m15;
      #pragma unroll
      for (int r = 0; r < 8; ++r) {
        float v = (acc[nt][r] + bb[n]) * (bb[64 + n] * BN_S) + bb[128 + n];
        h1w[(mb8 + r) * 64 + n] = (__bf16)fmaxf(v, 0.f);
      }
    }
    v8f acc2[4] = {};
    #pragma unroll
    for (int ks = 0; ks < 2; ++ks) {
      const __bf16* ap = h1w + m15 * 64 + ks * 32 + koff;
      v16bf a = ldfrag(ap, ap + 16);
      #pragma unroll
      for (int nt = 0; nt < 4; ++nt) {
        const __bf16* wp = w2 + (ks * 64 + nt * 16 + m15) * 32 + half * 16;
        v16bf bf = ldfrag(wp, wp + 8);
        acc2[nt] = __builtin_amdgcn_wmma_f32_16x16x32_bf16(
            false, a, false, bf, (short)0, acc2[nt], false, false);
      }
    }
    #pragma unroll
    for (int nt = 0; nt < 4; ++nt) {
      int n = nt * 16 + m15;
      #pragma unroll
      for (int r = 0; r < 8; ++r)
        h2[(ebase + mb8 + r) * 64 + n] = acc2[nt][r] + bb[192 + n];
    }
  }
  __syncthreads();

  // max over 20 edges per point; write f32 (next kNN), bf16 (next conv), Hcat
  for (int o = tid; o < NPTS * 64; o += 256) {
    int p = o >> 6, c = o & 63;
    float m = -3.0e38f;
    #pragma unroll
    for (int k = 0; k < KNN; ++k) m = fmaxf(m, h2[(p * KNN + k) * 64 + c]);
    size_t row = (size_t)b * NPT + pbase + p;
    Xf[row * 64 + c] = m;
    Xh[row * 64 + c] = (__bf16)m;
    Hcat[row * 192 + hoff + c] = (__bf16)m;
  }
}

// ---------------------------------------------------------------------------
// Head GEMM: 8 waves/block, each wave a 16x64 tile of relu(A@W + b).
// W pre-swizzled so every fragment is two 16B global loads.
// ---------------------------------------------------------------------------
template <bool RELU>
__global__ __launch_bounds__(256) void k_gemm(const __bf16* __restrict__ A,
                                              const __bf16* __restrict__ Ws,
                                              const float* __restrict__ bias,
                                              __bf16* __restrict__ Out,
                                              int Kdim, int Nc) {
  const int lane = threadIdx.x & 31, wave = threadIdx.x >> 5;
  const int mbase = blockIdx.x * 16;
  const int nbase = (blockIdx.y * 8 + wave) * 64;
  if (nbase >= Nc) return;
  const int m15 = lane & 15, half = lane >> 4;
  const int mb8 = half * 8, koff = half * 8;
  v8f acc[4] = {};
  for (int ks = 0; ks < Kdim / 32; ++ks) {
    const __bf16* ap = A + (size_t)(mbase + m15) * Kdim + ks * 32 + koff;
    v16bf a = ldfrag(ap, ap + 16);
    #pragma unroll
    for (int nt = 0; nt < 4; ++nt) {
      const __bf16* wp =
          Ws + (size_t)(ks * Nc + nbase + nt * 16 + m15) * 32 + half * 16;
      v16bf bf = ldfrag(wp, wp + 8);
      acc[nt] = __builtin_amdgcn_wmma_f32_16x16x32_bf16(
          false, a, false, bf, (short)0, acc[nt], false, false);
    }
  }
  #pragma unroll
  for (int nt = 0; nt < 4; ++nt) {
    int n = nbase + nt * 16 + m15;
    #pragma unroll
    for (int r = 0; r < 8; ++r) {
      float v = acc[nt][r] + bias[n];
      if (RELU) v = fmaxf(v, 0.f);
      Out[(size_t)(mbase + mb8 + r) * Nc + n] = (__bf16)v;
    }
  }
}

// ---------------------------------------------------------------------------
// Final 128->40 layer + log_softmax, one thread per point (tiny).
// ---------------------------------------------------------------------------
__global__ __launch_bounds__(128) void k_final(const __bf16* __restrict__ A,
                                               const float* __restrict__ W,
                                               const float* __restrict__ bias,
                                               float* __restrict__ Out) {
  int row = blockIdx.x * blockDim.x + threadIdx.x;
  if (row >= BATCH * NPT) return;
  float o[40];
  #pragma unroll
  for (int n = 0; n < 40; ++n) o[n] = bias[n];
  for (int k = 0; k < 128; ++k) {
    float a = (float)A[(size_t)row * 128 + k];
    #pragma unroll
    for (int n = 0; n < 40; ++n) o[n] += a * W[k * 40 + n];
  }
  float mx = o[0];
  #pragma unroll
  for (int n = 1; n < 40; ++n) mx = fmaxf(mx, o[n]);
  float se = 0.f;
  #pragma unroll
  for (int n = 0; n < 40; ++n) se += __expf(o[n] - mx);
  float ls = __logf(se) + mx;
  #pragma unroll
  for (int n = 0; n < 40; ++n) Out[(size_t)row * 40 + n] = o[n] - ls;
}

// ---------------------------------------------------------------------------
extern "C" void kernel_launch(void* const* d_in, const int* in_sizes, int n_in,
                              void* d_out, int out_size, void* d_ws,
                              size_t ws_size, hipStream_t stream) {
  (void)in_sizes; (void)n_in; (void)out_size; (void)ws_size;
  const float* x   = (const float*)d_in[0];
  const float* pos = (const float*)d_in[1];
  const float* cw[3][6];
  for (int i = 0; i < 3; ++i)
    for (int j = 0; j < 6; ++j) cw[i][j] = (const float*)d_in[2 + i * 6 + j];
  const float* mW1 = (const float*)d_in[20]; const float* mb1 = (const float*)d_in[21];
  const float* mW2 = (const float*)d_in[22]; const float* mb2 = (const float*)d_in[23];
  const float* mW3 = (const float*)d_in[24]; const float* mb3 = (const float*)d_in[25];
  const float* mW4 = (const float*)d_in[26]; const float* mb4 = (const float*)d_in[27];

  char* ws = (char*)d_ws;
  size_t off = 0;
  auto take = [&](size_t bytes) {
    char* p = ws + off;
    off = (off + bytes + 255) & ~(size_t)255;
    return p;
  };
  const size_t R = (size_t)BATCH * NPT;  // 32768 points
  float*  F0f = (float*) take(R * 8 * 4);
  __bf16* F0h = (__bf16*)take(R * 8 * 2);
  int*    idx = (int*)   take(R * KNN * 4);
  float*  X1f = (float*) take(R * 64 * 4);
  __bf16* X1h = (__bf16*)take(R * 64 * 2);
  float*  X2f = (float*) take(R * 64 * 4);
  __bf16* X2h = (__bf16*)take(R * 64 * 2);
  float*  X3f = (float*) take(R * 64 * 4);
  __bf16* X3h = (__bf16*)take(R * 64 * 2);
  __bf16* Hc  = (__bf16*)take(R * 192 * 2);
  __bf16* W1b = (__bf16*)take((size_t)192 * 1024 * 2);
  __bf16* W2b = (__bf16*)take((size_t)1024 * 256 * 2);
  __bf16* W3b = (__bf16*)take((size_t)256 * 128 * 2);
  __bf16* A1  = (__bf16*)take(R * 1024 * 2);
  __bf16* A2  = (__bf16*)take(R * 256 * 2);
  __bf16* A3  = (__bf16*)take(R * 128 * 2);

  k_concat<<<dim3((unsigned)((R + 255) / 256)), 256, 0, stream>>>(x, pos, F0f, F0h);

  const size_t knn_lds = (size_t)(16 * NPT + NPT + 16) * 4;       // ~272KB
  const size_t lds_e1 = (size_t)(320 * 32 + 32 * 64 + 64 * 64 + 8 * 16 * 64) * 2
                      + (size_t)(320 * 64 + 4 * 64) * 4 + 320 * 4;  // ~130KB
  const size_t lds_e2 = (size_t)(320 * 128 + 128 * 64 + 64 * 64 + 8 * 16 * 64) * 2
                      + (size_t)(320 * 64 + 4 * 64) * 4 + 320 * 4;  // ~202KB

  dim3 gk(NPT / 16, BATCH);
  k_knn<6, 8><<<gk, 256, knn_lds, stream>>>(F0f, idx);
  k_edge<6, 8, 32><<<gk, 256, lds_e1, stream>>>(
      F0h, idx, cw[0][0], cw[0][1], cw[0][2], cw[0][3], cw[0][4], cw[0][5],
      X1f, X1h, Hc, 0);
  k_knn<64, 64><<<gk, 256, knn_lds, stream>>>(X1f, idx);
  k_edge<64, 64, 128><<<gk, 256, lds_e2, stream>>>(
      X1h, idx, cw[1][0], cw[1][1], cw[1][2], cw[1][3], cw[1][4], cw[1][5],
      X2f, X2h, Hc, 64);
  k_knn<64, 64><<<gk, 256, knn_lds, stream>>>(X2f, idx);
  k_edge<64, 64, 128><<<gk, 256, lds_e2, stream>>>(
      X2h, idx, cw[2][0], cw[2][1], cw[2][2], cw[2][3], cw[2][4], cw[2][5],
      X3f, X3h, Hc, 128);

  k_swz<<<(192 * 1024 + 255) / 256, 256, 0, stream>>>(mW1, W1b, 192, 1024);
  k_swz<<<(1024 * 256 + 255) / 256, 256, 0, stream>>>(mW2, W2b, 1024, 256);
  k_swz<<<(256 * 128 + 255) / 256, 256, 0, stream>>>(mW3, W3b, 256, 128);

  k_gemm<true><<<dim3((unsigned)(R / 16), 2), 256, 0, stream>>>(Hc, W1b, mb1, A1, 192, 1024);
  k_gemm<true><<<dim3((unsigned)(R / 16), 1), 256, 0, stream>>>(A1, W2b, mb2, A2, 1024, 256);
  k_gemm<true><<<dim3((unsigned)(R / 16), 1), 256, 0, stream>>>(A2, W3b, mb3, A3, 256, 128);

  k_final<<<dim3((unsigned)((R + 127) / 128)), 128, 0, stream>>>(A3, mW4, mb4, (float*)d_out);
}
